// BaselineAttn_42545946034825
// MI455X (gfx1250) — matrix operand order repaired and checked by reading, not compiled, not run on hardware
//
#include <hip/hip_runtime.h>

#define BB 4
#define TT 2048
#define EE 1024
#define HH 16
#define DHd 64
#define RR 8

typedef __attribute__((ext_vector_type(16))) __bf16          v16bf;
typedef __attribute__((ext_vector_type(8)))  float           v8f;
typedef __attribute__((ext_vector_type(8)))  unsigned short  v8us;
typedef __attribute__((ext_vector_type(16))) unsigned short  v16us;

__device__ __forceinline__ unsigned short f2bf(float f) {
    unsigned int u = __builtin_bit_cast(unsigned int, f);
    u += 0x7FFFu + ((u >> 16) & 1u);          // round-to-nearest-even
    return (unsigned short)(u >> 16);
}

// Build a 16x bf16 fragment from two 16-byte chunks (LDS or global).
__device__ __forceinline__ v16bf make_frag(const unsigned short* p0,
                                           const unsigned short* p1) {
    v8us a = *(const v8us*)p0;
    v8us b = *(const v8us*)p1;
    v16us c = __builtin_shufflevector(a, b, 0,1,2,3,4,5,6,7,8,9,10,11,12,13,14,15);
    return __builtin_bit_cast(v16bf, c);
}

// CDNA5 async global->LDS copy (16B per lane), tracked by ASYNCcnt.
__device__ __forceinline__ void async_g2l_b128(unsigned lds_off, const void* gptr) {
    asm volatile("global_load_async_to_lds_b128 %0, %1, off"
                 :: "v"(lds_off), "v"(gptr) : "memory");
}
#define S_WAIT_ASYNCCNT(n) asm volatile("s_wait_asynccnt " #n ::: "memory")

__device__ __forceinline__ unsigned lds_off(const void* p) {
    return (unsigned)(unsigned long long)p;   // low 32 bits of flat LDS addr = LDS offset
}

// ---------------------------------------------------------------- converts
__global__ void cvt_bf16_k(const float* __restrict__ in,
                           unsigned short* __restrict__ out, int n) {
    int i = blockIdx.x * blockDim.x + threadIdx.x;
    if (i < n) out[i] = f2bf(in[i]);
}

// W_eff[n,k] = W[n,k] + 2 * sum_t A[n,t]*Bm[t,k]   (LoRA folded), bf16 out
__global__ void weff_k(const float* __restrict__ W, const float* __restrict__ A,
                       const float* __restrict__ Bm, unsigned short* __restrict__ out) {
    int i = blockIdx.x * blockDim.x + threadIdx.x;   // over E*E
    int n = i >> 10;
    int k = i & 1023;
    float acc = W[i];
#pragma unroll
    for (int t = 0; t < RR; ++t)
        acc += 2.0f * A[n * RR + t] * Bm[t * EE + k];
    out[i] = f2bf(acc);
}

// ---------------------------------------------------------------- GEMM
// C[M,N] = A[M,K] * W[N,K]^T   (bf16 in, f32 accum, bf16 or f32 out)
// block tile 128x128, 8 waves (4 in M x 2 in N), wave tile 32x64, K chunk 64.
// Double-buffered LDS fed by async global->LDS loads (prefetch distance 2).
__global__ __launch_bounds__(256)
void gemm_bf16_k(const unsigned short* __restrict__ A,
                 const unsigned short* __restrict__ W,
                 void* __restrict__ C, int M, int N, int K, int bf16out) {
    __shared__ __align__(16) unsigned short As[2][128 * 72];
    __shared__ __align__(16) unsigned short Bs[2][128 * 72];

    const int tid  = threadIdx.x;
    const int wid  = tid >> 5;
    const int lane = tid & 31;
    const int l    = lane & 15;
    const int hi   = lane >> 4;
    const int Mb   = blockIdx.x * 128;
    const int Nb   = blockIdx.y * 128;
    const int mw   = (wid & 3) * 32;
    const int nw   = (wid >> 2) * 64;

    // issue the 8 async b128 loads this thread owns for one 64-wide K-chunk
    auto issue = [&](int kc, int buf) {
#pragma unroll
        for (int s = tid; s < 1024; s += 256) {  // 128 rows x 8 b128 per tile
            int r = s >> 3, c = (s & 7) * 8;
            async_g2l_b128(lds_off(&As[buf][r * 72 + c]),
                           &A[(size_t)(Mb + r) * K + kc + c]);
            async_g2l_b128(lds_off(&Bs[buf][r * 72 + c]),
                           &W[(size_t)(Nb + r) * K + kc + c]);
        }
    };

    v8f acc[2][4];
#pragma unroll
    for (int mt = 0; mt < 2; ++mt)
#pragma unroll
        for (int nt = 0; nt < 4; ++nt)
#pragma unroll
            for (int e = 0; e < 8; ++e) acc[mt][nt][e] = 0.0f;

    issue(0, 0);
    if (64 < K) issue(64, 1);

    int it = 0;
    for (int kc = 0; kc < K; kc += 64, ++it) {
        const int buf = it & 1;
        // in-order retire: <=8 outstanding => this chunk's 8 are complete
        S_WAIT_ASYNCCNT(0x8);
        __syncthreads();

#pragma unroll
        for (int kk = 0; kk < 2; ++kk) {           // two 32-wide K slices
            v16bf af[2], bfm[4];
#pragma unroll
            for (int mt = 0; mt < 2; ++mt) {       // A layout: K interleave @8
                int row = mw + mt * 16 + l;
                af[mt] = make_frag(&As[buf][row * 72 + kk * 32 + hi * 8],
                                   &As[buf][row * 72 + kk * 32 + 16 + hi * 8]);
            }
#pragma unroll
            for (int nt = 0; nt < 4; ++nt) {       // B layout: K interleave @8
                int row = nw + nt * 16 + l;
                bfm[nt] = make_frag(&Bs[buf][row * 72 + kk * 32 + hi * 8],
                                    &Bs[buf][row * 72 + kk * 32 + 16 + hi * 8]);
            }
#pragma unroll
            for (int mt = 0; mt < 2; ++mt)
#pragma unroll
                for (int nt = 0; nt < 4; ++nt)
                    acc[mt][nt] = __builtin_amdgcn_wmma_f32_16x16x32_bf16(
                        false, af[mt], false, bfm[nt], (short)0, acc[mt][nt], false, false);
        }

        __syncthreads();                            // everyone done reading buf
        if (kc + 128 < K) issue(kc + 128, buf);     // refill just-freed buffer
    }

#pragma unroll
    for (int mt = 0; mt < 2; ++mt)
#pragma unroll
        for (int nt = 0; nt < 4; ++nt)
#pragma unroll
            for (int e = 0; e < 8; ++e) {
                int row = Mb + mw + mt * 16 + e + hi * 8;
                int col = Nb + nw + nt * 16 + l;
                float v = acc[mt][nt][e];
                if (bf16out) ((unsigned short*)C)[(size_t)row * N + col] = f2bf(v);
                else         ((float*)C)[(size_t)row * N + col] = v;
            }
}

// ---------------------------------------------------------------- attention
// One block: 128 q rows for one (b,h). 8 waves, one 16-row q tile each.
// Flash-style online softmax with ALiBi + causal mask; WMMA for QK^T and PV.
// K tiles: async global->LDS, double buffered; V: register prefetch (dist 2)
// then transpose-scatter into double-buffered LDS.
__global__ __launch_bounds__(256)
void attn_k(const unsigned short* __restrict__ Q,
            const unsigned short* __restrict__ Kg,
            const unsigned short* __restrict__ Vg,
            unsigned short* __restrict__ O) {
    __shared__ __align__(16) unsigned short Ks[2][32 * 72];    // [key][dh]
    __shared__ __align__(16) unsigned short Vt[2][64 * 40];    // [dh][key]
    __shared__ __align__(16) unsigned short Ps[8 * 16 * 40];   // per-wave P tile

    const int tid  = threadIdx.x;
    const int wid  = tid >> 5;
    const int lane = tid & 31;
    const int l    = lane & 15;
    const int hi   = lane >> 4;
    const int qb   = blockIdx.x * 128;
    const int h    = blockIdx.y;
    const int b    = blockIdx.z;
    const int qw   = qb + wid * 16;
    const size_t baseRow = (size_t)b * TT;
    const int hoff = h * DHd;
    const float slope = exp2f(-0.5f * (float)(h + 1));   // ALiBi, H=16

    const int skey = tid >> 3;            // staging: key row this thread owns
    const int scol = (tid & 7) * 8;       // staging: dh column (x8)

    auto issueK = [&](int kc, int buf) {
        async_g2l_b128(lds_off(&Ks[buf][skey * 72 + scol]),
                       &Kg[(baseRow + kc + skey) * EE + hoff + scol]);
    };
    auto loadV = [&](int kc) -> v8us {
        return *(const v8us*)&Vg[(baseRow + kc + skey) * EE + hoff + scol];
    };

    // Q fragments for the two dh-halves (A layout, direct from global)
    v16bf qf[2];
    {
        const unsigned short* qr = &Q[(baseRow + qw + l) * EE + hoff];
        qf[0] = make_frag(qr + hi * 8,      qr + 16 + hi * 8);
        qf[1] = make_frag(qr + 32 + hi * 8, qr + 48 + hi * 8);
    }

    float m[8], ssum[8];
    v8f o[4];
#pragma unroll
    for (int e = 0; e < 8; ++e) { m[e] = -3.0e38f; ssum[e] = 0.0f; }
#pragma unroll
    for (int d = 0; d < 4; ++d)
#pragma unroll
        for (int e = 0; e < 8; ++e) o[d][e] = 0.0f;

    const int kend = qb + 128;                 // causal upper bound for block
    v8us vv[2];
    vv[0] = loadV(0);  issueK(0, 0);           // kend >= 128: both chunks valid
    vv[1] = loadV(32); issueK(32, 1);

    int it = 0;
    for (int kc = 0; kc < kend; kc += 32, ++it) {
        const int buf = it & 1;
        // transpose-scatter this chunk's V registers into Vt[buf]
#pragma unroll
        for (int j = 0; j < 8; ++j) Vt[buf][(scol + j) * 40 + skey] = vv[buf][j];
        // in-order retire: <=1 outstanding => this chunk's K tile is in LDS
        S_WAIT_ASYNCCNT(0x1);
        __syncthreads();

        if (kc <= qw + 15) {                   // wave-uniform causal skip
            // ---- scores: S[16q x 32key], contract over dh=64 (2 steps)
            v8f sc[2];
#pragma unroll
            for (int t = 0; t < 2; ++t) {
#pragma unroll
                for (int e = 0; e < 8; ++e) sc[t][e] = 0.0f;
#pragma unroll
                for (int st = 0; st < 2; ++st) {
                    const unsigned short* kr =
                        &Ks[buf][(t * 16 + l) * 72 + st * 32 + hi * 8];
                    v16bf kf = make_frag(kr, kr + 16);
                    sc[t] = __builtin_amdgcn_wmma_f32_16x16x32_bf16(
                        false, qf[st], false, kf, (short)0, sc[t], false, false);
                }
            }
            // ---- online softmax (rows live in element index, keys in lanes)
            unsigned short* pb = &Ps[wid * 640];
#pragma unroll
            for (int e = 0; e < 8; ++e) {
                int i  = qw + e + hi * 8;
                int j0 = kc + l;
                int j1 = kc + 16 + l;
                float s0 = sc[0][e] * 0.125f + slope * (float)(i - j0);
                float s1 = sc[1][e] * 0.125f + slope * (float)(i - j1);
                if (j0 > i) s0 = -1.0e30f;
                if (j1 > i) s1 = -1.0e30f;
                float mc = fmaxf(s0, s1);
#pragma unroll
                for (int msk = 1; msk < 16; msk <<= 1)
                    mc = fmaxf(mc, __shfl_xor(mc, msk, 32));
                float mn = fmaxf(m[e], mc);
                float p0 = __expf(s0 - mn);
                float p1 = __expf(s1 - mn);
                float rs = p0 + p1;
#pragma unroll
                for (int msk = 1; msk < 16; msk <<= 1)
                    rs += __shfl_xor(rs, msk, 32);
                float al = __expf(m[e] - mn);
                m[e] = mn;
                ssum[e] = ssum[e] * al + rs;
#pragma unroll
                for (int d = 0; d < 4; ++d) o[d][e] *= al;
                pb[(e + hi * 8) * 40 + l]      = f2bf(p0);
                pb[(e + hi * 8) * 40 + 16 + l] = f2bf(p1);
            }
            // ---- PV: A = P[16x32] (from per-wave LDS), B = V^T tiles
            v16bf pf = make_frag(&pb[l * 40 + hi * 8], &pb[l * 40 + 16 + hi * 8]);
#pragma unroll
            for (int dt = 0; dt < 4; ++dt) {
                const unsigned short* vr = &Vt[buf][(dt * 16 + l) * 40 + hi * 8];
                v16bf vf = make_frag(vr, vr + 16);
                o[dt] = __builtin_amdgcn_wmma_f32_16x16x32_bf16(
                    false, pf, false, vf, (short)0, o[dt], false, false);
            }
        }

        __syncthreads();                        // everyone done reading buf
        if (kc + 64 < kend) {                   // prefetch distance 2
            vv[buf] = loadV(kc + 64);
            issueK(kc + 64, buf);
        }
    }

#pragma unroll
    for (int dt = 0; dt < 4; ++dt)
#pragma unroll
        for (int e = 0; e < 8; ++e) {
            int row = qw + e + hi * 8;
            float v = o[dt][e] / ssum[e];
            O[(baseRow + row) * EE + hoff + dt * 16 + l] = f2bf(v);
        }
}

// ---------------------------------------------------------------- launch
extern "C" void kernel_launch(void* const* d_in, const int* in_sizes, int n_in,
                              void* d_out, int out_size, void* d_ws, size_t ws_size,
                              hipStream_t stream) {
    const float* x  = (const float*)d_in[0];
    const float* Wq = (const float*)d_in[1];
    const float* Aq = (const float*)d_in[2];
    const float* Bq = (const float*)d_in[3];
    const float* Wk = (const float*)d_in[4];
    const float* Ak = (const float*)d_in[5];
    const float* Bk = (const float*)d_in[6];
    const float* Wv = (const float*)d_in[7];
    const float* Av = (const float*)d_in[8];
    const float* Bv = (const float*)d_in[9];
    const float* Wp = (const float*)d_in[10];

    const size_t XN = (size_t)BB * TT * EE;   // 8,388,608
    const size_t WN = (size_t)EE * EE;        // 1,048,576

    unsigned short* ws  = (unsigned short*)d_ws;
    unsigned short* xb  = ws;            // x bf16
    unsigned short* wqb = xb  + XN;      // W_eff q
    unsigned short* wkb = wqb + WN;
    unsigned short* wvb = wkb + WN;
    unsigned short* wpb = wvb + WN;      // Wp bf16
    unsigned short* qbf = wpb + WN;      // q bf16
    unsigned short* kbf = qbf + XN;
    unsigned short* vbf = kbf + XN;
    unsigned short* abf = vbf + XN;      // attention out bf16

    cvt_bf16_k<<<(int)((XN + 255) / 256), 256, 0, stream>>>(x,  xb,  (int)XN);
    cvt_bf16_k<<<(int)((WN + 255) / 256), 256, 0, stream>>>(Wp, wpb, (int)WN);
    weff_k<<<(int)(WN / 256), 256, 0, stream>>>(Wq, Aq, Bq, wqb);
    weff_k<<<(int)(WN / 256), 256, 0, stream>>>(Wk, Ak, Bk, wkb);
    weff_k<<<(int)(WN / 256), 256, 0, stream>>>(Wv, Av, Bv, wvb);

    dim3 gg(BB * TT / 128, EE / 128);
    gemm_bf16_k<<<gg, 256, 0, stream>>>(xb, wqb, qbf, BB * TT, EE, EE, 1);
    gemm_bf16_k<<<gg, 256, 0, stream>>>(xb, wkb, kbf, BB * TT, EE, EE, 1);
    gemm_bf16_k<<<gg, 256, 0, stream>>>(xb, wvb, vbf, BB * TT, EE, EE, 1);

    attn_k<<<dim3(TT / 128, HH, BB), 256, 0, stream>>>(qbf, kbf, vbf, abf);

    gemm_bf16_k<<<gg, 256, 0, stream>>>(abf, wpb, d_out, BB * TT, EE, EE, 0);
}
